// ResumeJobGNN_38362647888476
// MI455X (gfx1250) — matrix operand order repaired and checked by reading, not compiled, not run on hardware
//
#include <hip/hip_runtime.h>
#include <hip/hip_bf16.h>
#include <math.h>

#define NS 10000
#define NP 2048
#define EJ 32768
#define ER 32768
#define GD 1536
#define GH 8

typedef __attribute__((ext_vector_type(16))) __bf16 v16bf;
typedef __attribute__((ext_vector_type(8)))  __bf16 v8bf;
typedef __attribute__((ext_vector_type(8)))  float  v8f;

__device__ __forceinline__ __bf16 f2bf(float f) {
  union { float f; unsigned u; } x; x.f = f;
  unsigned r = x.u + 0x7FFFu + ((x.u >> 16) & 1u);
  unsigned short h = (unsigned short)(r >> 16);
  union { unsigned short s; __bf16 b; } y; y.s = h;
  return y.b;
}

__device__ __forceinline__ float wred_sum(float v) {
#pragma unroll
  for (int m = 16; m > 0; m >>= 1) v += __shfl_xor(v, m, 32);
  return v;
}
__device__ __forceinline__ float wred_max(float v) {
#pragma unroll
  for (int m = 16; m > 0; m >>= 1) v = fmaxf(v, __shfl_xor(v, m, 32));
  return v;
}

// low 32 bits of a generic pointer to LDS == DS byte offset (ISA 10.2 aperture)
__device__ __forceinline__ unsigned lds_off32(const void* p) {
  return (unsigned)(unsigned long long)p;
}

// CDNA5 async global->LDS 16B copy, tracked by ASYNCcnt
__device__ __forceinline__ void async_copy_b128(unsigned lds_byte_addr, const void* gptr) {
  asm volatile("global_load_async_to_lds_b128 %0, %1, off"
               :: "v"(lds_byte_addr), "v"(gptr) : "memory");
}
__device__ __forceinline__ void async_wait0() {
  asm volatile("s_wait_asynccnt 0x0" ::: "memory");
}

// ---------------------------------------------------------------------------
// bf16 WMMA GEMM with async double-buffered LDS staging.
//   C[M,N] = alphaScale*(A[M,K] @ Bt[N,K]^T) [+bias] [+=C] [leaky]
// A: bf16 row-major [M,K]; Bt: bf16 row-major [N,K] (pre-transposed weights)
// Requires M%128==0, N%128==0, K%32==0 (all call sites satisfy this).
// ---------------------------------------------------------------------------
#define GF_ACC   1
#define GF_BIAS  2
#define GF_LEAKY 4

__global__ __launch_bounds__(256)
void gemm_bf16_wmma(const __bf16* __restrict__ A, const __bf16* __restrict__ Bt,
                    int K, float* __restrict__ C, int N,
                    const float* __restrict__ bias, float alphaScale, int flags)
{
  // padded K-stride (40 bf16 = 80B, multiple of 16B) for conflict-spread b128 access
  __shared__ __attribute__((aligned(16))) __bf16 As[2][128 * 40];
  __shared__ __attribute__((aligned(16))) __bf16 Bs[2][128 * 40];

  const int tid = threadIdx.x;
  const int m0 = blockIdx.y * 128;
  const int n0 = blockIdx.x * 128;

  const int wid = tid >> 5, lane = tid & 31;
  const int wm = wid & 3, wn = wid >> 2;      // 4 waves in M, 2 in N
  const int l15 = lane & 15;
  const bool lo = lane < 16;
  const int abase = lo ? 0 : 8;               // A frag: K base per lane half
  const int bbase = lo ? 0 : 8;               // B frag: K base per lane half (same as A)

  // per-thread static copy slots: 512 16B chunks per tile, 2 chunks/thread/tile
  const int c0 = tid * 2, c1 = tid * 2 + 1;
  const int r0 = c0 >> 2, q0 = c0 & 3;        // row 0..127, 16B chunk 0..3
  const int r1 = c1 >> 2, q1 = c1 & 3;

  auto issue_tile = [&](int k0, __bf16* as, __bf16* bs) {
    async_copy_b128(lds_off32(&as[r0 * 40 + q0 * 8]), A  + (size_t)(m0 + r0) * K + k0 + q0 * 8);
    async_copy_b128(lds_off32(&as[r1 * 40 + q1 * 8]), A  + (size_t)(m0 + r1) * K + k0 + q1 * 8);
    async_copy_b128(lds_off32(&bs[r0 * 40 + q0 * 8]), Bt + (size_t)(n0 + r0) * K + k0 + q0 * 8);
    async_copy_b128(lds_off32(&bs[r1 * 40 + q1 * 8]), Bt + (size_t)(n0 + r1) * K + k0 + q1 * 8);
  };

  v8f acc[2][4] = {};

  issue_tile(0, As[0], Bs[0]);
  async_wait0();
  __syncthreads();

  const int nk = K >> 5;
  for (int kb = 0; kb < nk; ++kb) {
    const int cur = kb & 1, nxt = cur ^ 1;
    if (kb + 1 < nk) issue_tile((kb + 1) * 32, As[nxt], Bs[nxt]); // overlap w/ WMMA

    const __bf16* as = As[cur];
    const __bf16* bs = Bs[cur];

    v16bf afrag[2], bfrag[4];
#pragma unroll
    for (int im = 0; im < 2; ++im) {
      int r = wm * 32 + im * 16 + l15;
      v8bf p0 = *(const v8bf*)&as[r * 40 + abase];        // K base..base+7
      v8bf p1 = *(const v8bf*)&as[r * 40 + 16 + abase];   // K 16+base..
#pragma unroll
      for (int i = 0; i < 8; ++i) { afrag[im][i] = p0[i]; afrag[im][i + 8] = p1[i]; }
    }
#pragma unroll
    for (int jn = 0; jn < 4; ++jn) {
      int c = wn * 64 + jn * 16 + l15;
      v8bf q0v = *(const v8bf*)&bs[c * 40 + bbase];
      v8bf q1v = *(const v8bf*)&bs[c * 40 + 16 + bbase];
#pragma unroll
      for (int i = 0; i < 8; ++i) { bfrag[jn][i] = q0v[i]; bfrag[jn][i + 8] = q1v[i]; }
    }
#pragma unroll
    for (int im = 0; im < 2; ++im)
#pragma unroll
      for (int jn = 0; jn < 4; ++jn)
        acc[im][jn] = __builtin_amdgcn_wmma_f32_16x16x32_bf16(
            false, afrag[im], false, bfrag[jn], (short)0, acc[im][jn], false, false);

    async_wait0();       // next tile landed
    __syncthreads();
  }

  // ---- epilogue: scale / bias / accumulate / leaky ----
#pragma unroll
  for (int im = 0; im < 2; ++im) {
#pragma unroll
    for (int jn = 0; jn < 4; ++jn) {
      int gc = n0 + wn * 64 + jn * 16 + l15;
      int gr0 = m0 + wm * 32 + im * 16 + (lo ? 0 : 8);
#pragma unroll
      for (int i = 0; i < 8; ++i) {
        int gr = gr0 + i;
        float v = acc[im][jn][i] * alphaScale;
        if (flags & GF_BIAS) v += bias[gc];
        if (flags & GF_ACC)  v += C[(size_t)gr * N + gc];
        if (flags & GF_LEAKY) v = v > 0.f ? v : 0.01f * v;
        C[(size_t)gr * N + gc] = v;
      }
    }
  }
}

// ---------------------------------------------------------------------------
// Weight prep: tiled transpose + f32->bf16.  out[n][h*K + k] = in[k][h*N + n]
// grid (K/32, N/32, H), block (32,8)
// ---------------------------------------------------------------------------
__global__ __launch_bounds__(256)
void k_transpose_bf16(const float* __restrict__ in, __bf16* __restrict__ out,
                      int K, int N) {
  __shared__ float t[32][33];
  const int H = gridDim.z;
  const int k0 = blockIdx.x * 32, n0 = blockIdx.y * 32, h = blockIdx.z;
  const int tx = threadIdx.x, ty = threadIdx.y;
#pragma unroll
  for (int i = 0; i < 4; ++i) {
    int kk = k0 + ty + i * 8;
    t[ty + i * 8][tx] = in[(size_t)kk * (H * N) + (size_t)h * N + n0 + tx];
  }
  __syncthreads();
#pragma unroll
  for (int i = 0; i < 4; ++i) {
    int nn = n0 + ty + i * 8;
    out[(size_t)nn * (H * K) + (size_t)h * K + k0 + tx] = f2bf(t[tx][ty + i * 8]);
  }
}

__global__ void k_cvt_bf16(const float* __restrict__ in, __bf16* __restrict__ out, int n) {
  for (int i = blockIdx.x * blockDim.x + threadIdx.x; i < n; i += gridDim.x * blockDim.x)
    out[i] = f2bf(in[i]);
}

// ---------------------------------------------------------------------------
// CSR construction (group edges by destination; removes atomics from hot path)
// ---------------------------------------------------------------------------
__global__ void k_zero_int(int* p, int n) {
  for (int i = blockIdx.x * blockDim.x + threadIdx.x; i < n; i += gridDim.x * blockDim.x)
    p[i] = 0;
}

__global__ void k_count(const int* __restrict__ dst, int* __restrict__ cnt, int E) {
  int e = blockIdx.x * 256 + threadIdx.x;
  if (e < E) atomicAdd(&cnt[dst[e]], 1);
}

__global__ __launch_bounds__(1024)
void k_scan(const int* __restrict__ cnt, int* __restrict__ rp) {
  __shared__ int s[2048];
  int t = threadIdx.x;
  s[t] = cnt[t]; s[t + 1024] = cnt[t + 1024];
  __syncthreads();
  for (int off = 1; off < 2048; off <<= 1) {
    int v0 = (t >= off) ? s[t - off] : 0;
    int v1 = ((t + 1024) >= off) ? s[t + 1024 - off] : 0;
    __syncthreads();
    s[t] += v0; s[t + 1024] += v1;
    __syncthreads();
  }
  rp[t] = s[t] - cnt[t];
  rp[t + 1024] = s[t + 1024] - cnt[t + 1024];
  if (t == 1023) rp[2048] = s[2047];
}

__global__ void k_fill(const int* __restrict__ dst, const int* __restrict__ rp,
                       int* __restrict__ cur, int* __restrict__ list, int E) {
  int e = blockIdx.x * 256 + threadIdx.x;
  if (e < E) { int d = dst[e]; int pos = atomicAdd(&cur[d], 1); list[rp[d] + pos] = e; }
}

// ---------------------------------------------------------------------------
// v_src[d,h] = W_gat_src[d, h*GD:..] . att_src[h]   (and same for dst side)
// ---------------------------------------------------------------------------
__global__ __launch_bounds__(256)
void k_vproj(const float* __restrict__ Wsrc, const float* __restrict__ attsrc,
             const float* __restrict__ Wdst, const float* __restrict__ attdst,
             float* __restrict__ vsrc, float* __restrict__ vdst) {
  int wave = blockIdx.x * 8 + (threadIdx.x >> 5);
  int lane = threadIdx.x & 31;
  if (wave >= 2 * GD * GH) return;
  int which = wave >= GD * GH;
  int r = wave - which * GD * GH;
  int d = r >> 3, h = r & 7;
  const float* W = which ? Wdst : Wsrc;
  const float* att = which ? attdst : attsrc;
  float s = 0.f;
  for (int j = lane; j < GD; j += 32)
    s += W[(size_t)d * (GH * GD) + h * GD + j] * att[h * GD + j];
  s = wred_sum(s);
  if (lane == 0) (which ? vdst : vsrc)[d * GH + h] = s;
}

// a_src[n,h] = x_skill[n] . v_src[:,h] ; a_dst[p,h] = x_resume[p] . v_dst[:,h]
__global__ __launch_bounds__(256)
void k_aproj(const float* __restrict__ x_skill, const float* __restrict__ x_resume,
             const float* __restrict__ vsrc, const float* __restrict__ vdst,
             float* __restrict__ a_src, float* __restrict__ a_dst) {
  int wave = blockIdx.x * 8 + (threadIdx.x >> 5);
  int lane = threadIdx.x & 31;
  if (wave >= NS + NP) return;
  const float *x, *v; float* out; int n;
  if (wave < NS) { x = x_skill;  v = vsrc; out = a_src; n = wave; }
  else           { x = x_resume; v = vdst; out = a_dst; n = wave - NS; }
  float acc[GH] = {};
  for (int d = lane; d < GD; d += 32) {
    float xv = x[(size_t)n * GD + d];
#pragma unroll
    for (int h = 0; h < GH; ++h) acc[h] += xv * v[d * GH + h];
  }
#pragma unroll
  for (int h = 0; h < GH; ++h) {
    float s = wred_sum(acc[h]);
    if (lane == 0) out[n * GH + h] = s;
  }
}

// segment softmax per destination (one wave per resume node), writes alpha [E,H]
__global__ __launch_bounds__(256)
void k_alpha(const int* __restrict__ rp, const int* __restrict__ list,
             const int* __restrict__ esrc,
             const float* __restrict__ a_src, const float* __restrict__ a_dst,
             float* __restrict__ alpha_out) {
  int p = blockIdx.x * 8 + (threadIdx.x >> 5);
  int lane = threadIdx.x & 31;
  if (p >= NP) return;
  int b = rp[p], eend = rp[p + 1];
#pragma unroll 1
  for (int h = 0; h < GH; ++h) {
    float ad = a_dst[p * GH + h];
    float m = -3.4e38f;
    for (int i = b + lane; i < eend; i += 32) {
      int s = esrc[list[i]];
      float v = a_src[s * GH + h] + ad;
      v = v > 0.f ? v : 0.2f * v;
      m = fmaxf(m, v);
    }
    m = wred_max(m);
    float sum = 0.f;
    for (int i = b + lane; i < eend; i += 32) {
      int s = esrc[list[i]];
      float v = a_src[s * GH + h] + ad;
      v = v > 0.f ? v : 0.2f * v;
      sum += __expf(v - m);
    }
    sum = wred_sum(sum);
    float inv = 1.f / (sum + 1e-16f);
    for (int i = b + lane; i < eend; i += 32) {
      int e = list[i]; int s = esrc[e];
      float v = a_src[s * GH + h] + ad;
      v = v > 0.f ? v : 0.2f * v;
      alpha_out[(size_t)e * GH + h] = __expf(v - m) * inv;
    }
  }
}

// GraphConv aggregation: agg[p] = sum_e x_skill[src_e]  (bf16 out, GEMM A operand)
__global__ __launch_bounds__(256)
void k_agg_job(const int* __restrict__ rp, const int* __restrict__ list,
               const int* __restrict__ esrc, const float* __restrict__ x_skill,
               __bf16* __restrict__ agg) {
  int p = blockIdx.x, tid = threadIdx.x;
  float r[6] = {};
  int b = rp[p], e = rp[p + 1];
  for (int i = b; i < e; ++i) {
    const float* xr = x_skill + (size_t)esrc[list[i]] * GD;
#pragma unroll
    for (int j = 0; j < 6; ++j) r[j] += xr[tid + j * 256];
  }
#pragma unroll
  for (int j = 0; j < 6; ++j) agg[(size_t)p * GD + tid + j * 256] = f2bf(r[j]);
}

// GAT aggregation of RAW features: agg[p,h,:] = sum_e alpha[e,h] * x_skill[src_e]
__global__ __launch_bounds__(256)
void k_agg_res(const int* __restrict__ rp, const int* __restrict__ list,
               const int* __restrict__ esrc, const float* __restrict__ x_skill,
               const float* __restrict__ alpha, __bf16* __restrict__ agg) {
  int p = blockIdx.x, tid = threadIdx.x;
  float r[GH][6] = {};
  int b = rp[p], e = rp[p + 1];
  for (int i = b; i < e; ++i) {
    int eid = list[i];
    const float* xr = x_skill + (size_t)esrc[eid] * GD;
    float al[GH];
#pragma unroll
    for (int h = 0; h < GH; ++h) al[h] = alpha[(size_t)eid * GH + h];
#pragma unroll
    for (int j = 0; j < 6; ++j) {
      float xv = xr[tid + j * 256];
#pragma unroll
      for (int h = 0; h < GH; ++h) r[h][j] += al[h] * xv;
    }
  }
#pragma unroll
  for (int h = 0; h < GH; ++h)
#pragma unroll
    for (int j = 0; j < 6; ++j)
      agg[(size_t)p * (GH * GD) + h * GD + tid + j * 256] = f2bf(r[h][j]);
}

// hcat = [leaky(h_job), leaky(h_res)] in bf16 (GEMM A operand for MLP layer 1)
__global__ void k_concat(const float* __restrict__ h_job, const float* __restrict__ h_res,
                         __bf16* __restrict__ hcat) {
  int total = NP * 2 * GD;
  for (int i = blockIdx.x * blockDim.x + threadIdx.x; i < total; i += gridDim.x * blockDim.x) {
    int p = i / (2 * GD), j = i % (2 * GD);
    float v = (j < GD) ? h_job[(size_t)p * GD + j] : h_res[(size_t)p * GD + (j - GD)];
    hcat[i] = f2bf(v > 0.f ? v : 0.01f * v);
  }
}

// score[p] = m2[p] . W3 + b3
__global__ __launch_bounds__(256)
void k_final(const float* __restrict__ m2, const float* __restrict__ W3,
             const float* __restrict__ b3, float* __restrict__ score) {
  int p = blockIdx.x * 8 + (threadIdx.x >> 5);
  int lane = threadIdx.x & 31;
  if (p >= NP) return;
  float s = 0.f;
#pragma unroll
  for (int j = 0; j < 4; ++j) s += m2[(size_t)p * 128 + lane + j * 32] * W3[lane + j * 32];
  s = wred_sum(s);
  if (lane == 0) score[p] = s + b3[0];
}

// ---------------------------------------------------------------------------
extern "C" void kernel_launch(void* const* d_in, const int* in_sizes, int n_in,
                              void* d_out, int out_size, void* d_ws, size_t ws_size,
                              hipStream_t stream) {
  const float* x_skill = (const float*)d_in[0];
  const float* x_job   = (const float*)d_in[1];
  const float* x_res   = (const float*)d_in[2];
  const float* W_rel   = (const float*)d_in[3];
  const float* b_rel   = (const float*)d_in[4];
  const float* W_root  = (const float*)d_in[5];
  const float* W_gs    = (const float*)d_in[6];
  const float* W_gd    = (const float*)d_in[7];
  const float* att_s   = (const float*)d_in[8];
  const float* att_d   = (const float*)d_in[9];
  const float* b_gat   = (const float*)d_in[10];
  const float* W1      = (const float*)d_in[11];
  const float* b1      = (const float*)d_in[12];
  const float* W2      = (const float*)d_in[13];
  const float* b2      = (const float*)d_in[14];
  const float* W3      = (const float*)d_in[15];
  const float* b3      = (const float*)d_in[16];
  const int* esj = (const int*)d_in[17];
  const int* edj = (const int*)d_in[18];
  const int* esr = (const int*)d_in[19];
  const int* edr = (const int*)d_in[20];

  // ---- workspace layout (~160 MB, float/bf16 buffers 16B-aligned) ----
  float* ws = (float*)d_ws;
  size_t off = 0;
  auto fal = [&](size_t n) { float* p = ws + off; off += n; return p; };
  float* h_job = fal((size_t)NP * GD);
  float* h_res = fal((size_t)NP * GD);
  float* m1    = fal((size_t)NP * 512);
  float* m2    = fal((size_t)NP * 128);
  float* v_src = fal((size_t)GD * GH);
  float* v_dst = fal((size_t)GD * GH);
  float* a_src = fal((size_t)NS * GH);
  float* a_dst = fal((size_t)NP * GH);

  __bf16* bp = (__bf16*)(ws + off);
  size_t boff = 0;
  auto bal = [&](size_t n) { __bf16* p = bp + boff; boff += n; return p; };
  __bf16* aggjob_b = bal((size_t)NP * GD);
  __bf16* xjob_b   = bal((size_t)NP * GD);
  __bf16* aggres_b = bal((size_t)NP * GH * GD);
  __bf16* hcat_b   = bal((size_t)NP * 2 * GD);
  __bf16* m1_b     = bal((size_t)NP * 512);
  __bf16* Wrel_t   = bal((size_t)GD * GD);
  __bf16* Wroot_t  = bal((size_t)GD * GD);
  __bf16* Wgat_t   = bal((size_t)GD * GH * GD);
  __bf16* W1_t     = bal((size_t)512 * 2 * GD);
  __bf16* W2_t     = bal((size_t)128 * 512);

  int* ip = (int*)(bp + boff);
  int* cnt_j = ip; ip += NP;
  int* cur_j = ip; ip += NP;
  int* rp_j  = ip; ip += NP + 1;
  int* list_j = ip; ip += EJ;
  int* cnt_r = ip; ip += NP;
  int* cur_r = ip; ip += NP;
  int* rp_r  = ip; ip += NP + 1;
  int* list_r = ip; ip += ER;

  float* score = (float*)d_out;
  float* alpha = score + NP;   // output alpha [ER, GH], also consumed by k_agg_res

  dim3 t32x8(32, 8);

  // --- weight prep: transpose+convert to bf16 [N,K] for async B staging ---
  k_transpose_bf16<<<dim3(GD / 32, GD / 32, 1), t32x8, 0, stream>>>(W_rel, Wrel_t, GD, GD);
  k_transpose_bf16<<<dim3(GD / 32, GD / 32, 1), t32x8, 0, stream>>>(W_root, Wroot_t, GD, GD);
  k_transpose_bf16<<<dim3(GD / 32, GD / 32, GH), t32x8, 0, stream>>>(W_gs, Wgat_t, GD, GD);
  k_transpose_bf16<<<dim3(2 * GD / 32, 512 / 32, 1), t32x8, 0, stream>>>(W1, W1_t, 2 * GD, 512);
  k_transpose_bf16<<<dim3(512 / 32, 128 / 32, 1), t32x8, 0, stream>>>(W2, W2_t, 512, 128);
  k_cvt_bf16<<<1024, 256, 0, stream>>>(x_job, xjob_b, NP * GD);

  // --- CSR build for both edge sets ---
  k_zero_int<<<16, 256, 0, stream>>>(cnt_j, 2 * NP);
  k_zero_int<<<16, 256, 0, stream>>>(cnt_r, 2 * NP);
  k_count<<<EJ / 256, 256, 0, stream>>>(edj, cnt_j, EJ);
  k_count<<<ER / 256, 256, 0, stream>>>(edr, cnt_r, ER);
  k_scan<<<1, 1024, 0, stream>>>(cnt_j, rp_j);
  k_scan<<<1, 1024, 0, stream>>>(cnt_r, rp_r);
  k_fill<<<EJ / 256, 256, 0, stream>>>(edj, rp_j, cur_j, list_j, EJ);
  k_fill<<<ER / 256, 256, 0, stream>>>(edr, rp_r, cur_r, list_r, ER);

  // --- GAT attention logits (algebraically reduced) + segment softmax ---
  k_vproj<<<(2 * GD * GH) / 8, 256, 0, stream>>>(W_gs, att_s, W_gd, att_d, v_src, v_dst);
  k_aproj<<<(NS + NP + 7) / 8, 256, 0, stream>>>(x_skill, x_res, v_src, v_dst, a_src, a_dst);
  k_alpha<<<NP / 8, 256, 0, stream>>>(rp_r, list_r, esr, a_src, a_dst, alpha);

  // --- aggregations (raw-feature, per-destination, atomic-free, bf16 out) ---
  k_agg_job<<<NP, 256, 0, stream>>>(rp_j, list_j, esj, x_skill, aggjob_b);
  k_agg_res<<<NP, 256, 0, stream>>>(rp_r, list_r, esr, x_skill, alpha, aggres_b);

  // --- WMMA GEMMs (async double-buffered LDS staging) ---
  // h_job = agg_job @ W_rel + b_rel
  gemm_bf16_wmma<<<dim3(GD / 128, NP / 128), 256, 0, stream>>>(
      aggjob_b, Wrel_t, GD, h_job, GD, b_rel, 1.f, GF_BIAS);
  // h_job += x_job @ W_root
  gemm_bf16_wmma<<<dim3(GD / 128, NP / 128), 256, 0, stream>>>(
      xjob_b, Wroot_t, GD, h_job, GD, nullptr, 1.f, GF_ACC);
  // h_res = (1/H) * agg_res @ Wgat + b_gat   (folds head projection + head-mean)
  gemm_bf16_wmma<<<dim3(GD / 128, NP / 128), 256, 0, stream>>>(
      aggres_b, Wgat_t, GH * GD, h_res, GD, b_gat, 1.f / GH, GF_BIAS);

  // --- MLP ---
  k_concat<<<2048, 256, 0, stream>>>(h_job, h_res, hcat_b);
  gemm_bf16_wmma<<<dim3(512 / 128, NP / 128), 256, 0, stream>>>(
      hcat_b, W1_t, 2 * GD, m1, 512, b1, 1.f, GF_BIAS | GF_LEAKY);
  k_cvt_bf16<<<1024, 256, 0, stream>>>(m1, m1_b, NP * 512);
  gemm_bf16_wmma<<<dim3(128 / 128, NP / 128), 256, 0, stream>>>(
      m1_b, W2_t, 512, m2, 128, b2, 1.f, GF_BIAS | GF_LEAKY);
  k_final<<<NP / 8, 256, 0, stream>>>(m2, W3, b3, score);
}
